// MambaRNN_66056597012622
// MI455X (gfx1250) — matrix operand order repaired and checked by reading, not compiled, not run on hardware
//
#include <hip/hip_runtime.h>

// ---------------------------------------------------------------------------
// Mamba forward for MI455X (gfx1250, wave32, WMMA).
// All GEMMs on v_wmma_f32_16x16x32_bf16. Weights are converted f32->bf16 ONCE
// per call (embedding included: 77MB bf16 stays L2-resident across M-tiles),
// so the GEMM inner loop is pure global_load_b128 + v_wmma.
// Per-wave tile: 32(M) x 64(N) -> 8 WMMAs per 32-wide K step.
// ---------------------------------------------------------------------------

typedef __attribute__((ext_vector_type(16))) __bf16         v16bf;
typedef __attribute__((ext_vector_type(8)))  float          v8f;
typedef __attribute__((ext_vector_type(8)))  unsigned short u16x8;

__device__ __forceinline__ unsigned short f2bf(float f) {
  unsigned u = __float_as_uint(f);
  u += 0x7FFFu + ((u >> 16) & 1u);   // round-to-nearest-even
  return (unsigned short)(u >> 16);
}

// ---------------- f32 -> bf16 streaming converter ----------------
__global__ __launch_bounds__(256) void k_cvt_bf16(
    const float* __restrict__ in, unsigned short* __restrict__ out, size_t n) {
  size_t i = (size_t)blockIdx.x * 256 + threadIdx.x;
  size_t stride = (size_t)gridDim.x * 256;
  for (; i < n; i += stride) out[i] = f2bf(in[i]);
}

// ---------------- embedding gather ----------------
__global__ __launch_bounds__(256) void k_embed_gather(
    const int* __restrict__ ids, const int* __restrict__ mask,
    const float* __restrict__ embed, float* __restrict__ h,
    int BL, int Dm) {
  int idx = blockIdx.x * 256 + threadIdx.x;
  if (idx >= BL * Dm) return;
  int t = idx / Dm, d = idx - t * Dm;
  h[idx] = embed[(size_t)ids[t] * Dm + d] * (float)mask[t];
}

// ---------------- RMSNorm -> bf16 (one block per token) ----------------
__global__ __launch_bounds__(256) void k_rmsnorm_bf16(
    const float* __restrict__ x, const float* __restrict__ w,
    unsigned short* __restrict__ out, int D) {
  __shared__ float red[8];
  int t = blockIdx.x;
  const float* xr = x + (size_t)t * D;
  float s = 0.f;
  for (int d = threadIdx.x; d < D; d += 256) { float v = xr[d]; s += v * v; }
  for (int o = 16; o > 0; o >>= 1) s += __shfl_xor(s, o, 32);
  if ((threadIdx.x & 31) == 0) red[threadIdx.x >> 5] = s;
  __syncthreads();
  if (threadIdx.x == 0) {
    float tot = 0.f;
    for (int i = 0; i < 8; ++i) tot += red[i];
    red[0] = rsqrtf(tot / (float)D + 1e-5f);
  }
  __syncthreads();
  float r = red[0];
  for (int d = threadIdx.x; d < D; d += 256)
    out[(size_t)t * D + d] = f2bf(xr[d] * r * w[d]);
}

// ---------------- WMMA GEMM: out[M,N] = A_bf16[M,K] * W_bf16[N,K]^T (+Cin) --
// One wave per block; wave owns a 32(M) x 64(N) tile: 8 WMMAs per K-step.
// Fragment layouts per CDNA5 ISA 7.12.2 (wave32):
//   A 16x32 bf16 : lane row = lane&15 ; e<8 -> K=k0+(lane>=16?8:0)+e ,
//                  e>=8 -> K=k0+16+(lane>=16?8:0)+(e-8)   (two 16B loads)
//   B 32x16 bf16 : lane col = lane&15 ; same K mapping as A (two 16B loads)
//   C/D f32      : vgpr d -> row d+8*(lane>=16), col lane&15
__global__ __launch_bounds__(32) void k_gemm_bf16(
    const unsigned short* __restrict__ A, const unsigned short* __restrict__ W,
    const float* __restrict__ Cin, float* __restrict__ out,
    int M, int N, int K) {
  const int lane = threadIdx.x & 31;
  const int half = lane >> 4;
  const int l15  = lane & 15;
  const int m0 = blockIdx.y << 5;   // 2 M-tiles
  const int n0 = blockIdx.x << 6;   // 4 N-tiles

  v8f acc[2][4];
#pragma unroll
  for (int mt = 0; mt < 2; ++mt)
#pragma unroll
    for (int t = 0; t < 4; ++t)
#pragma unroll
      for (int e = 0; e < 8; ++e) acc[mt][t][e] = 0.f;

  const unsigned short* arow0 = A + (size_t)(m0 + l15) * K;
  const unsigned short* arow1 = arow0 + (size_t)16 * K;

  union Frag { u16x8 h[2]; v16bf v; };

  for (int k0 = 0; k0 < K; k0 += 32) {
    Frag a0, a1;
    a0.h[0] = *(const u16x8*)(arow0 + k0 + half * 8);
    a0.h[1] = *(const u16x8*)(arow0 + k0 + 16 + half * 8);
    a1.h[0] = *(const u16x8*)(arow1 + k0 + half * 8);
    a1.h[1] = *(const u16x8*)(arow1 + k0 + 16 + half * 8);
#pragma unroll
    for (int t = 0; t < 4; ++t) {
      int n  = n0 + t * 16 + l15;
      int nc = n < N ? n : N - 1;             // clamp loads; store is guarded
      const unsigned short* wrow = W + (size_t)nc * K + k0 + half * 8;
      Frag b;
      b.h[0] = *(const u16x8*)(wrow);
      b.h[1] = *(const u16x8*)(wrow + 16);
      acc[0][t] = __builtin_amdgcn_wmma_f32_16x16x32_bf16(
          false, a0.v, false, b.v, (short)0, acc[0][t], false, false);
      acc[1][t] = __builtin_amdgcn_wmma_f32_16x16x32_bf16(
          false, a1.v, false, b.v, (short)0, acc[1][t], false, false);
    }
  }

#pragma unroll
  for (int mt = 0; mt < 2; ++mt) {
#pragma unroll
    for (int t = 0; t < 4; ++t) {
      int n = n0 + t * 16 + l15;
      if (n < N) {
#pragma unroll
        for (int d = 0; d < 8; ++d) {
          size_t idx = (size_t)(m0 + mt * 16 + d + 8 * half) * N + n;
          float v = acc[mt][t][d];
          if (Cin) v += Cin[idx];
          out[idx] = v;
        }
      }
    }
  }
}

// ---------------- causal depthwise conv (K=4) + SiLU ----------------
__global__ __launch_bounds__(256) void k_conv_silu(
    const float* __restrict__ xz, const float* __restrict__ cw,
    const float* __restrict__ cb, float* __restrict__ xo,
    unsigned short* __restrict__ xo_bf, int L, int Di, int total) {
  int idx = blockIdx.x * 256 + threadIdx.x;
  if (idx >= total) return;
  int d = idx % Di;
  int t = idx / Di;             // t = b*L + l
  int l = t % L;
  float acc = cb[d];
#pragma unroll
  for (int k = 0; k < 4; ++k) {
    int lk = l + k - 3;
    if (lk >= 0) acc += cw[d * 4 + k] * xz[(size_t)(t + k - 3) * 3072 + d];
  }
  float s = acc / (1.f + __expf(-acc));     // silu
  xo[idx] = s;
  xo_bf[idx] = f2bf(s);
}

// ---------------- dt = softplus(dt_r @ dt_w^T + dt_b) ----------------
__global__ __launch_bounds__(256) void k_dt_softplus(
    const float* __restrict__ proj, const float* __restrict__ dtw,
    const float* __restrict__ dtb, float* __restrict__ dt,
    int Di, int total) {
  int idx = blockIdx.x * 256 + threadIdx.x;
  if (idx >= total) return;
  int d = idx % Di;
  int t = idx / Di;
  const float* p  = proj + (size_t)t * 80;        // dt_r = proj[:,0:48]
  const float* wv = dtw + (size_t)d * 48;
  float a = dtb[d];
#pragma unroll 8
  for (int r = 0; r < 48; ++r) a += p[r] * wv[r];
  dt[idx] = (a > 20.f) ? a : log1pf(__expf(a));   // softplus
}

// ---------------- selective scan, fused gate + bf16 out ----------------
// block = 256 threads = 16 channels x 16 states; grid = B * Di/16.
// Each thread runs one (b,d,n) recurrence; Ns=16 reduction via shfl_xor.
__global__ __launch_bounds__(256) void k_scan(
    const float* __restrict__ dt, const float* __restrict__ proj,
    const float* __restrict__ x, const float* __restrict__ xz,
    const float* __restrict__ A_log, const float* __restrict__ Dp,
    unsigned short* __restrict__ y_bf, int L, int Di) {
  int n  = threadIdx.x & 15;
  int dg = threadIdx.x >> 4;
  int blocksPerB = Di >> 4;
  int b = blockIdx.x / blocksPerB;
  int d = (blockIdx.x % blocksPerB) * 16 + dg;
  float A   = -__expf(A_log[d * 16 + n]);
  float Dpv = Dp[d];
  float h = 0.f;
  for (int l = 0; l < L; ++l) {
    size_t t = (size_t)b * L + l;
    float dtv = dt[t * Di + d];
    float Bv  = proj[t * 80 + 48 + n];   // Bm = proj[:,48:64]
    float Cv  = proj[t * 80 + 64 + n];   // Cm = proj[:,64:80]
    float xv  = x[t * Di + d];
    h = __expf(dtv * A) * h + dtv * Bv * xv;
    float c = h * Cv;
    c += __shfl_xor(c, 1, 32);
    c += __shfl_xor(c, 2, 32);
    c += __shfl_xor(c, 4, 32);
    c += __shfl_xor(c, 8, 32);
    if (n == 0) {
      float y = c + Dpv * xv;
      float z = xz[t * 3072 + 1536 + d];
      y *= z / (1.f + __expf(-z));       // y * silu(z)
      y_bf[t * Di + d] = f2bf(y);
    }
  }
}

// ---------------------------------------------------------------------------
extern "C" void kernel_launch(void* const* d_in, const int* in_sizes, int n_in,
                              void* d_out, int out_size, void* d_ws, size_t ws_size,
                              hipStream_t stream) {
  (void)in_sizes; (void)n_in; (void)out_size; (void)ws_size;
  const int B = 2, L = 512, BL = 1024, Dm = 768, Di = 1536, NL = 4, V = 50280;

  const int*   q_ids  = (const int*)d_in[0];
  const int*   q_mask = (const int*)d_in[1];
  const float* embed  = (const float*)d_in[4];
  const float* in_w   = (const float*)d_in[5];   // (NL, 3072, 768)
  const float* conv_w = (const float*)d_in[6];   // (NL, 1536, 4)
  const float* conv_b = (const float*)d_in[7];   // (NL, 1536)
  const float* x_w    = (const float*)d_in[8];   // (NL, 80, 1536)
  const float* dt_w   = (const float*)d_in[9];   // (NL, 1536, 48)
  const float* dt_b   = (const float*)d_in[10];  // (NL, 1536)
  const float* A_log  = (const float*)d_in[11];  // (NL, 1536, 16)
  const float* Dp     = (const float*)d_in[12];  // (NL, 1536)
  const float* out_w  = (const float*)d_in[13];  // (NL, 768, 1536)
  const float* norm_w = (const float*)d_in[14];  // (NL, 768)
  const float* fnorm  = (const float*)d_in[15];  // (768)

  // ---- carve workspace (~125 MB; embedding bf16 copy is the big item) ----
  char* p = (char*)d_ws;
  auto carve = [&](size_t bytes) -> void* {
    void* r = (void*)p; p += (bytes + 255) & ~(size_t)255; return r;
  };
  unsigned short* embed_bf = (unsigned short*)carve((size_t)V * Dm * 2);
  unsigned short* w_in_bf  = (unsigned short*)carve((size_t)2 * Di * Dm * 2);
  unsigned short* w_x_bf   = (unsigned short*)carve((size_t)80 * Di * 2);
  unsigned short* w_out_bf = (unsigned short*)carve((size_t)Dm * Di * 2);
  float*          h     = (float*)carve((size_t)BL * Dm * 4);
  unsigned short* r_bf  = (unsigned short*)carve((size_t)BL * Dm * 2);
  float*          xz    = (float*)carve((size_t)BL * 2 * Di * 4);
  float*          xf    = (float*)carve((size_t)BL * Di * 4);
  unsigned short* x_bf  = (unsigned short*)carve((size_t)BL * Di * 2);
  float*          proj  = (float*)carve((size_t)BL * 80 * 4);
  float*          dt    = (float*)carve((size_t)BL * Di * 4);
  unsigned short* y_bf  = (unsigned short*)carve((size_t)BL * Di * 2);
  unsigned short* hn_bf = (unsigned short*)carve((size_t)BL * Dm * 2);

  // ---- one-time f32->bf16 weight conversions (amortized over the GEMMs) ----
  k_cvt_bf16<<<4096, 256, 0, stream>>>(embed, embed_bf, (size_t)V * Dm);

  // ---- embed + mask ----
  {
    int tot = BL * Dm;
    k_embed_gather<<<(tot + 255) / 256, 256, 0, stream>>>(q_ids, q_mask, embed, h, BL, Dm);
  }

  for (int l = 0; l < NL; ++l) {
    // per-layer weight conversions (stream-ordered before their GEMMs)
    k_cvt_bf16<<<2048, 256, 0, stream>>>(in_w + (size_t)l * 2 * Di * Dm, w_in_bf,
                                         (size_t)2 * Di * Dm);
    k_cvt_bf16<<<512, 256, 0, stream>>>(x_w + (size_t)l * 80 * Di, w_x_bf,
                                        (size_t)80 * Di);
    k_cvt_bf16<<<1024, 256, 0, stream>>>(out_w + (size_t)l * Dm * Di, w_out_bf,
                                         (size_t)Dm * Di);

    // rmsnorm -> bf16
    k_rmsnorm_bf16<<<BL, 256, 0, stream>>>(h, norm_w + (size_t)l * Dm, r_bf, Dm);

    // in_proj: xz[BL,3072] = r @ in_w^T
    {
      dim3 g((2 * Di + 63) / 64, BL / 32);
      k_gemm_bf16<<<g, 32, 0, stream>>>(r_bf, w_in_bf, nullptr, xz, BL, 2 * Di, Dm);
    }

    // depthwise conv + silu
    {
      int tot = BL * Di;
      k_conv_silu<<<(tot + 255) / 256, 256, 0, stream>>>(
          xz, conv_w + (size_t)l * Di * 4, conv_b + (size_t)l * Di, xf, x_bf, L, Di, tot);
    }

    // x_proj: proj[BL,80] = x @ x_w^T
    {
      dim3 g((80 + 63) / 64, BL / 32);
      k_gemm_bf16<<<g, 32, 0, stream>>>(x_bf, w_x_bf, nullptr, proj, BL, 80, Di);
    }

    // dt = softplus(dt_r @ dt_w^T + dt_b)
    {
      int tot = BL * Di;
      k_dt_softplus<<<(tot + 255) / 256, 256, 0, stream>>>(
          proj, dt_w + (size_t)l * Di * 48, dt_b + (size_t)l * Di, dt, Di, tot);
    }

    // selective scan + gate -> y_bf
    k_scan<<<B * (Di / 16), 256, 0, stream>>>(
        dt, proj, xf, xz, A_log + (size_t)l * Di * 16, Dp + (size_t)l * Di, y_bf, L, Di);

    // out_proj with residual: h += y @ out_w^T
    {
      dim3 g((Dm + 63) / 64, BL / 32);
      k_gemm_bf16<<<g, 32, 0, stream>>>(y_bf, w_out_bf, h, h, BL, Dm, Di);
    }
  }

  // final rmsnorm -> bf16
  k_rmsnorm_bf16<<<BL, 256, 0, stream>>>(h, fnorm, hn_bf, Dm);

  // logits: out[BL,V] = hn @ embed^T  (77MB bf16 embed is L2-resident)
  {
    dim3 g((V + 63) / 64, BL / 32);
    k_gemm_bf16<<<g, 32, 0, stream>>>(hn_bf, embed_bf, nullptr, (float*)d_out, BL, V, Dm);
  }
}
